// MMAF_14877766713891
// MI455X (gfx1250) — matrix operand order repaired and checked by reading, not compiled, not run on hardware
//
#include <hip/hip_runtime.h>
#include <math.h>

// ---------------------------------------------------------------------------
// MMAF fusion block for MI455X (gfx1250, wave32, WMMA).
// Every dense contraction runs through one templated bf16-WMMA GEMM
// (v_wmma_f32_16x16x32_bf16, f32 accumulate) with parametrized address maps:
//   AMODE 0: affine (handles all token/CHW transposes for free)  [tile 64x32]
//   AMODE 1: implicit im2col 3x3 pad1 (offset convs)             [tile 32x64]
//   AMODE 2: deformable bilinear sampling, LDS-staged tap table  [tile 32x64]
// Gather modes use an N-wide tile so each K-step samples only 32 expensive
// A elements per 8 WMMAs. Branchless inner loops: OOB rows/cols are merely
// address-clamped (their results are never stored), K-tail masking happens
// once in a uniform tail block (only the AGG=42 GEMMs). 32-bit addresses.
// ---------------------------------------------------------------------------

typedef __attribute__((ext_vector_type(16))) __bf16 v16bf;
typedef __attribute__((ext_vector_type(8)))  float  v8f;

#define BB    64
#define CC    768
#define NN    256
#define ECC   128
#define AGGN  42
#define AGGP  48
#define BCr   (BB*CC)   // 49152
#define BNr   (BB*NN)   // 16384
#define CAT_C 1536
#define CAT_BSTRIDE (CAT_C*NN)  // 393216
#define CAT_HALF    (CC*NN)     // 196608

struct GemmArgs {
  const float* A;
  const float* Bw;     // weights, row-major [N, b_col], k stride 1
  const float* bias;
  float* out;
  const float* wtab;   // deform tap weights [B,9,256,4]
  const int*   itab;   // deform tap indices [B,9,256,4]
  int M, N, K;
  int a_div, a_outer, a_inner, a_k;   // affine A map
  int x_b, x_c, x_pix;                // image strides (amode 1/2)
  int b_col;
  int o_div, o_outer, o_inner, o_col; // output map
  int epi;                            // 0 none, 1 relu, 2 sigmoid
};

__device__ __forceinline__ int koff(int i) { return (i < 8) ? i : 8 + i; }

__device__ __forceinline__ void fill16(v16bf& f, float4 q0, float4 q1,
                                       float4 q2, float4 q3) {
  f[0]=(__bf16)q0.x;  f[1]=(__bf16)q0.y;  f[2]=(__bf16)q0.z;  f[3]=(__bf16)q0.w;
  f[4]=(__bf16)q1.x;  f[5]=(__bf16)q1.y;  f[6]=(__bf16)q1.z;  f[7]=(__bf16)q1.w;
  f[8]=(__bf16)q2.x;  f[9]=(__bf16)q2.y;  f[10]=(__bf16)q2.z; f[11]=(__bf16)q2.w;
  f[12]=(__bf16)q3.x; f[13]=(__bf16)q3.y; f[14]=(__bf16)q3.z; f[15]=(__bf16)q3.w;
}

// Wave computes an (MT*16) x (NT*16) tile, MT*NT == 8 WMMAs per K-step.
template <int AMODE, bool AVEC, bool BVEC, int MT, int NT>
__global__ __launch_bounds__(128) void wmma_gemm(GemmArgs g) {
  constexpr int ROWS = 16 * MT;
  const int lane = threadIdx.x & 31;
  const int wv   = threadIdx.x >> 5;
  const int hl   = lane >> 4;
  const int ml   = lane & 15;
  const int m0   = blockIdx.y * ROWS;
  const int kbA  = hl * 8;
  const int kbB  = hl * 8;

  // ---- deform: stage the (b, ROWS-pixel) slice of the tap table in LDS ----
  __shared__ float s_w[(AMODE == 2) ? 9 * ROWS * 4 : 1];
  __shared__ int   s_i[(AMODE == 2) ? 9 * ROWS * 4 : 1];
  if constexpr (AMODE == 2) {
    const int b  = m0 >> 8;
    const int p0 = m0 & 255;
    for (int idx = threadIdx.x; idx < 9 * ROWS; idx += 128) {
      int kk = idx / ROWS, j = idx % ROWS;
      int gt = ((b * 9 + kk) * 256 + p0 + j) * 4;
      *(float4*)&s_w[idx * 4] = *(const float4*)(g.wtab + gt);
      *(int4*)&s_i[idx * 4]   = *(const int4*)(g.itab + gt);
    }
    __syncthreads();
  }

  const int nt0 = (blockIdx.x * 4 + wv) * NT;
  if (nt0 * 16 >= g.N) return;          // uniform whole-wave exit
  const int n0 = nt0 * 16;

  // B columns: clamp for address safety; OOB columns are never stored.
  int cb[NT];
#pragma unroll
  for (int nt = 0; nt < NT; ++nt) {
    int c = n0 + nt * 16 + ml;
    cb[nt] = ((c < g.N) ? c : g.N - 1) * g.b_col;
  }

  // Per-sub A bases (block-uniform div: a_div/o_div are multiples of ROWS).
  int rowbase[MT], bimg[MT], py_[MT], px_[MT], tloc[MT];
  if constexpr (AMODE == 0) {
    const int q = (m0 / g.a_div) * g.a_outer + (m0 % g.a_div) * g.a_inner;
#pragma unroll
    for (int sub = 0; sub < MT; ++sub)
      rowbase[sub] = q + (sub * 16 + ml) * g.a_inner;
  } else {
    const int b = m0 >> 8;
#pragma unroll
    for (int sub = 0; sub < MT; ++sub) {
      int pix = (m0 & 255) + sub * 16 + ml;   // stays inside one image
      bimg[sub] = b * g.x_b;
      py_[sub] = pix >> 4;
      px_[sub] = pix & 15;
      tloc[sub] = sub * 16 + ml;              // local pixel in LDS slice
    }
  }

  v8f acc[NT][MT] = {};
  const int Kfull = g.K & ~31;

  for (int k0 = 0; k0 < Kfull; k0 += 32) {
    if constexpr (AMODE == 0) {
      if (k0 + 32 < Kfull)
        __builtin_prefetch(g.A + rowbase[0] + (k0 + 32) * g.a_k, 0, 1);
    }

    // decode k -> (ci, kernel-pos) once; shared by all M-tiles
    int kci[16], kdy[16], kdx[16], kqk[16];
    if constexpr (AMODE != 0) {
#pragma unroll
      for (int i = 0; i < 16; ++i) {
        int k = k0 + kbA + koff(i);
        int ci = k / 9;
        int kk = k - ci * 9;
        kci[i] = ci; kqk[i] = kk;
        kdy[i] = kk / 3 - 1; kdx[i] = kk % 3 - 1;
      }
    }

    v16bf bf[NT];
    if constexpr (BVEC) {
#pragma unroll
      for (int nt = 0; nt < NT; ++nt) {
        const float4* p = (const float4*)(g.Bw + cb[nt] + k0 + kbB);
        fill16(bf[nt], p[0], p[1], p[4], p[5]);
      }
    } else {
#pragma unroll
      for (int nt = 0; nt < NT; ++nt) {
        const float* p = g.Bw + cb[nt] + k0 + kbB;
#pragma unroll
        for (int i = 0; i < 16; ++i) bf[nt][i] = (__bf16)p[koff(i)];
      }
    }

    v16bf af[MT];
    if constexpr (AMODE == 0) {
      if constexpr (AVEC) {
#pragma unroll
        for (int sub = 0; sub < MT; ++sub) {
          const float* base = g.A + rowbase[sub] + k0 + kbA;
          fill16(af[sub], *(const float4*)base, *(const float4*)(base + 4),
                 *(const float4*)(base + 16), *(const float4*)(base + 20));
        }
      } else {
#pragma unroll
        for (int sub = 0; sub < MT; ++sub)
#pragma unroll
          for (int i = 0; i < 16; ++i)
            af[sub][i] =
                (__bf16)g.A[rowbase[sub] + (k0 + kbA + koff(i)) * g.a_k];
      }
    } else if constexpr (AMODE == 1) {
#pragma unroll
      for (int sub = 0; sub < MT; ++sub)
#pragma unroll
        for (int i = 0; i < 16; ++i) {
          int py = py_[sub] + kdy[i];
          int px = px_[sub] + kdx[i];
          bool inb = ((unsigned)py < 16u) & ((unsigned)px < 16u);
          // py,px in [-1,16]: &15 keeps the address in-range; value masked.
          float v = g.A[bimg[sub] + kci[i] * g.x_c +
                        (((py & 15) << 4) + (px & 15)) * g.x_pix];
          af[sub][i] = (__bf16)(inb ? v : 0.f);
        }
    } else {  // AMODE == 2: 4-tap bilinear sample, taps from LDS
#pragma unroll
      for (int sub = 0; sub < MT; ++sub)
#pragma unroll
        for (int i = 0; i < 16; ++i) {
          int t = (kqk[i] * ROWS + tloc[sub]) * 4;
          float4 w = *(const float4*)&s_w[t];
          int4  id = *(const int4*)&s_i[t];
          int base = bimg[sub] + kci[i] * g.x_c;
          float s = w.x * g.A[base + id.x * g.x_pix]
                  + w.y * g.A[base + id.y * g.x_pix]
                  + w.z * g.A[base + id.z * g.x_pix]
                  + w.w * g.A[base + id.w * g.x_pix];
          af[sub][i] = (__bf16)s;
        }
    }

#pragma unroll
    for (int nt = 0; nt < NT; ++nt)
#pragma unroll
      for (int sub = 0; sub < MT; ++sub)
        acc[nt][sub] = __builtin_amdgcn_wmma_f32_16x16x32_bf16(
            false, af[sub], false, bf[nt], (short)0, acc[nt][sub], false,
            false);
  }

  // ---- K tail (K%32 != 0 happens only for affine AGG=42 GEMMs) ----
  if constexpr (AMODE == 0) {
    if (g.K & 31) {
      const int k0 = Kfull;
      v16bf bf[NT], af[MT];
#pragma unroll
      for (int nt = 0; nt < NT; ++nt)
#pragma unroll
        for (int i = 0; i < 16; ++i) {
          int k = k0 + kbB + koff(i);
          bool kv = k < g.K;
          bf[nt][i] = (__bf16)(kv ? g.Bw[cb[nt] + k] : 0.f);
        }
#pragma unroll
      for (int sub = 0; sub < MT; ++sub)
#pragma unroll
        for (int i = 0; i < 16; ++i) {
          int k = k0 + kbA + koff(i);
          bool kv = k < g.K;
          af[sub][i] =
              (__bf16)(kv ? g.A[rowbase[sub] + k * g.a_k] : 0.f);
        }
#pragma unroll
      for (int nt = 0; nt < NT; ++nt)
#pragma unroll
        for (int sub = 0; sub < MT; ++sub)
          acc[nt][sub] = __builtin_amdgcn_wmma_f32_16x16x32_bf16(
              false, af[sub], false, bf[nt], (short)0, acc[nt][sub], false,
              false);
    }
  }

  // ---- epilogue: bias + activation, generalized output map ----
  const int obase =
      (m0 / g.o_div) * g.o_outer + (m0 % g.o_div) * g.o_inner;
#pragma unroll
  for (int nt = 0; nt < NT; ++nt) {
    int col = n0 + nt * 16 + ml;
    if (col < g.N) {
      float bv = g.bias ? g.bias[col] : 0.f;
      int colterm = obase + col * g.o_col;
#pragma unroll
      for (int sub = 0; sub < MT; ++sub)
#pragma unroll
        for (int r = 0; r < 8; ++r) {
          float v = acc[nt][sub][r] + bv;
          if (g.epi == 1)      v = fmaxf(v, 0.f);
          else if (g.epi == 2) v = 1.f / (1.f + __expf(-v));
          g.out[colterm + (sub * 16 + hl * 8 + r) * g.o_inner] = v;
        }
    }
  }
}

// offsets [B,18,256] -> bilinear tap tables (torchvision DeformConv2d rules)
__global__ void build_tab(const float* __restrict__ off,
                          float* __restrict__ wtab, int* __restrict__ itab) {
  int t = blockIdx.x * blockDim.x + threadIdx.x;
  if (t >= BB * 9 * NN) return;
  int pix = t & 255;
  int r   = t >> 8;
  int kk  = r % 9;
  int b   = r / 9;
  int ob  = b * (18 * NN);
  float dy = off[ob + (2 * kk) * NN + pix];
  float dx = off[ob + (2 * kk + 1) * NN + pix];
  float py = (float)(pix >> 4) + (float)(kk / 3 - 1) + dy;
  float px = (float)(pix & 15) + (float)(kk % 3 - 1) + dx;
  float y0 = floorf(py), x0 = floorf(px);
  int o = t * 4;
#pragma unroll
  for (int j = 0; j < 4; ++j) {
    float yi = y0 + (float)(j >> 1);
    float xi = x0 + (float)(j & 1);
    float w = (1.f - fabsf(py - yi)) * (1.f - fabsf(px - xi));
    bool valid = (yi >= 0.f) && (yi < 16.f) && (xi >= 0.f) && (xi < 16.f);
    float yc = fminf(fmaxf(yi, 0.f), 15.f);
    float xc = fminf(fmaxf(xi, 0.f), 15.f);
    itab[o + j] = (int)yc * 16 + (int)xc;
    wtab[o + j] = valid ? w : 0.f;
  }
}

__global__ void ew_add(const float4* __restrict__ a,
                       const float4* __restrict__ b, float4* __restrict__ o,
                       int n4) {
  int t = blockIdx.x * blockDim.x + threadIdx.x;
  if (t < n4) {
    float4 x = a[t], y = b[t];
    o[t] = float4{x.x + y.x, x.y + y.y, x.z + y.z, x.w + y.w};
  }
}

// fused = g*ar + (1-g)*at, written in [B, N, C] token layout.
__global__ void gated_combine(const float* __restrict__ gbuf,
                              const float* __restrict__ cat,
                              float* __restrict__ out, int n4) {
  int t4 = blockIdx.x * blockDim.x + threadIdx.x;
  if (t4 >= n4) return;
  int t = t4 * 4;
  int c = t % CC;
  int p = t / CC;
  int b = p >> 8, pix = p & 255;
  int base = b * CAT_BSTRIDE + c * NN + pix;
  float4 gg = *(const float4*)(gbuf + t);
  float4 r;
  r.x = gg.x * cat[base]          + (1.f - gg.x) * cat[base + CAT_HALF];
  r.y = gg.y * cat[base + NN]     + (1.f - gg.y) * cat[base + NN + CAT_HALF];
  r.z = gg.z * cat[base + 2 * NN] + (1.f - gg.z) * cat[base + 2 * NN + CAT_HALF];
  r.w = gg.w * cat[base + 3 * NN] + (1.f - gg.w) * cat[base + 3 * NN + CAT_HALF];
  *(float4*)(out + t) = r;
}

// ---------------------------------------------------------------------------
// Host orchestration
// ---------------------------------------------------------------------------
static GemmArgs mk(const float* A, const float* Bw, const float* bias,
                   float* out, int M, int N, int K) {
  GemmArgs g{};
  g.A = A; g.Bw = Bw; g.bias = bias; g.out = out;
  g.M = M; g.N = N; g.K = K;
  g.a_div = 1 << 30; g.a_outer = 0; g.a_inner = K; g.a_k = 1;
  g.b_col = K;
  g.o_div = 1 << 30; g.o_outer = 0; g.o_inner = N; g.o_col = 1;
  g.epi = 0;
  return g;
}

static void run(const GemmArgs& g, int amode, hipStream_t s) {
  int ntiles = (g.N + 15) / 16;
  if (amode == 0) {
    // affine: 64x32 wave tile (cheap A loads, reuse B x4)
    dim3 grid((unsigned)((ntiles + 7) / 8), (unsigned)((g.M + 63) / 64));
    bool avec = (g.a_k == 1) && (g.a_inner % 4 == 0);
    bool bvec = (g.b_col % 4) == 0;
    if (avec && bvec) wmma_gemm<0, true,  true,  4, 2><<<grid, 128, 0, s>>>(g);
    else if (avec)    wmma_gemm<0, true,  false, 4, 2><<<grid, 128, 0, s>>>(g);
    else if (bvec)    wmma_gemm<0, false, true,  4, 2><<<grid, 128, 0, s>>>(g);
    else              wmma_gemm<0, false, false, 4, 2><<<grid, 128, 0, s>>>(g);
  } else {
    // gather modes: 32x64 wave tile (halve expensive A sampling per WMMA)
    dim3 grid((unsigned)((ntiles + 15) / 16), (unsigned)((g.M + 31) / 32));
    if (amode == 1) wmma_gemm<1, false, true, 2, 4><<<grid, 128, 0, s>>>(g);
    else            wmma_gemm<2, false, true, 2, 4><<<grid, 128, 0, s>>>(g);
  }
}

struct Img { const float* p; int Ci; int sb, sc, sp; };

extern "C" void kernel_launch(void* const* d_in, const int* in_sizes, int n_in,
                              void* d_out, int out_size, void* d_ws,
                              size_t ws_size, hipStream_t stream) {
  (void)in_sizes; (void)n_in; (void)out_size; (void)ws_size;
  const float* rgb = (const float*)d_in[0];
  const float* tir = (const float*)d_in[1];
  // params flattened as JAX pytree leaves (dict keys sorted, tuples in order):
  //  2..13 : align_rgb  3x{db,dw,ob,ow}
  // 14..25 : align_tir  3x{db,dw,ob,ow}
  // 26 g1b, 27 g1w, 28 g2b, 29 g2w
  // 30..35 : ir_align  (dW,db,mW,mb,uW,ub)
  // 36..41 : proj      (dW,db,mW,mb,uW,ub)
  // 42..47 : rgb_align (dW,db,mW,mb,uW,ub)
  auto P = [&](int i) -> const float* { return (const float*)d_in[i]; };

  size_t pos = 0;
  auto alloc = [&](size_t nfloats) -> float* {
    float* p = (float*)((char*)d_ws + pos);
    pos += ((nfloats * sizeof(float) + 255) & ~(size_t)255);
    return p;
  };
  float* r_al = alloc((size_t)BNr * CC);
  float* t_al = alloc((size_t)BNr * CC);
  float* r_p  = alloc((size_t)BNr * CC);
  float* t_p  = alloc((size_t)BNr * CC);
  float* h1a  = alloc((size_t)BCr * AGGP);
  float* h2a  = alloc((size_t)BCr * AGGP);
  float* h1p  = alloc((size_t)BNr * ECC);
  float* h2p  = alloc((size_t)BNr * ECC);
  float* offb = alloc((size_t)BB * 18 * NN);
  float* wtab = alloc((size_t)BB * 9 * NN * 4);
  int*   itab = (int*)alloc((size_t)BB * 9 * NN * 4);
  float* x1   = alloc((size_t)BB * ECC * NN);
  float* x2   = alloc((size_t)BB * ECC * NN);
  float* cat  = alloc((size_t)BB * CAT_C * NN);
  float* ghid = alloc((size_t)BNr * 384);
  float* gbuf = alloc((size_t)BNr * CC);
  float* rt    = r_al;   // r_al dead after rgb proj-down GEMM
  float* f_tok = t_al;   // t_al dead after tir proj-down GEMM

  // ---- token-axis align adapters (rows = (b,c), K over N) ----
  auto align_adapter = [&](const float* tok, int base, float* outTok) {
    GemmArgs g = mk(tok, P(base + 0), P(base + 1), h1a, BCr, AGGN, NN);
    g.a_div = CC; g.a_outer = NN * CC; g.a_inner = 1; g.a_k = CC;
    g.o_inner = AGGP;
    run(g, 0, stream);
    g = mk(h1a, P(base + 2), P(base + 3), h2a, BCr, AGGN, AGGN);
    g.a_inner = AGGP; g.o_inner = AGGP;
    run(g, 0, stream);
    g = mk(h2a, P(base + 4), P(base + 5), outTok, BCr, NN, AGGN);
    g.a_inner = AGGP;
    g.o_div = CC; g.o_outer = NN * CC; g.o_inner = 1; g.o_col = CC;
    run(g, 0, stream);
  };
  align_adapter(rgb, 42, r_al);
  align_adapter(tir, 30, t_al);

  // ---- channel projection adapters (rows = (b,n), K over C) ----
  auto proj_adapter = [&](const float* inTok, float* outTok) {
    GemmArgs g = mk(inTok, P(36), P(37), h1p, BNr, ECC, CC); run(g, 0, stream);
    g = mk(h1p, P(38), P(39), h2p, BNr, ECC, ECC);           run(g, 0, stream);
    g = mk(h2p, P(40), P(41), outTok, BNr, CC, ECC);         run(g, 0, stream);
  };
  proj_adapter(r_al, r_p);
  proj_adapter(t_al, t_p);
  {
    int n4 = BNr * CC / 4;
    ew_add<<<(n4 + 255) / 256, 256, 0, stream>>>(
        (const float4*)r_p, (const float4*)t_p, (float4*)rt, n4);
  }
  proj_adapter(rt, f_tok);

  // ---- deform stage: offset conv -> tap tables -> deform conv ----
  auto stage = [&](Img x, Img y, int base, float* outp, int Co, int o_outer) {
    GemmArgs g = mk(y.p, P(base + 3), P(base + 2), offb, BNr, 18, y.Ci * 9);
    g.x_b = y.sb; g.x_c = y.sc; g.x_pix = y.sp;
    g.o_div = NN; g.o_outer = 18 * NN; g.o_inner = 1; g.o_col = NN;
    run(g, 1, stream);
    int nt = BB * 9 * NN;
    build_tab<<<(nt + 255) / 256, 256, 0, stream>>>(offb, wtab, itab);
    g = mk(x.p, P(base + 1), P(base + 0), outp, BNr, Co, x.Ci * 9);
    g.x_b = x.sb; g.x_c = x.sc; g.x_pix = x.sp;
    g.wtab = wtab; g.itab = itab;
    g.o_div = NN; g.o_outer = o_outer; g.o_inner = 1; g.o_col = NN;
    run(g, 2, stream);
  };

  auto branch = [&](const float* xTok, int base0, float* catHalf) {
    Img x0{xTok,  CC,  NN * CC,  1,  CC};   // token layout
    Img yf{f_tok, CC,  NN * CC,  1,  CC};
    Img i1{x1,    ECC, ECC * NN, NN, 1};    // CHW layout
    Img i2{x2,    ECC, ECC * NN, NN, 1};
    stage(x0, yf, base0 + 0, x1, ECC, ECC * NN);
    stage(i1, i1, base0 + 4, x2, ECC, ECC * NN);
    stage(i2, i2, base0 + 8, catHalf, CC, CAT_BSTRIDE);
  };
  branch(r_p, 2,  cat);              // ar -> channels [0,768)
  branch(t_p, 14, cat + CAT_HALF);   // at -> channels [768,1536)

  // ---- gated fusion: 1x1 convs as GEMMs over pixels ----
  {
    GemmArgs g = mk(cat, P(27), P(26), ghid, BNr, 384, CAT_C);
    g.a_div = NN; g.a_outer = CAT_BSTRIDE; g.a_inner = 1; g.a_k = NN;
    g.epi = 1;                         // ReLU
    run(g, 0, stream);
    g = mk(ghid, P(29), P(28), gbuf, BNr, CC, 384);
    g.epi = 2;                         // sigmoid
    run(g, 0, stream);
    int n4 = BNr * CC / 4;
    gated_combine<<<(n4 + 255) / 256, 256, 0, stream>>>(gbuf, cat,
                                                        (float*)d_out, n4);
  }
}
